// GNN_31722628448327
// MI455X (gfx1250) — matrix operand order repaired and checked by reading, not compiled, not run on hardware
//
#include <hip/hip_runtime.h>
#include <hip/hip_bf16.h>

typedef __attribute__((ext_vector_type(16))) _Float16 v16h;
typedef __attribute__((ext_vector_type(8)))  float    v8f;
typedef __attribute__((ext_vector_type(4)))  float    v4f;

#define N_NODES  100000
#define N_EDGES  3200000
#define N_GRAPHS 512
#define HID      32

// ---------------------------------------------------------------------------
// Generic zero-fill
// ---------------------------------------------------------------------------
__global__ void k_zero(float* __restrict__ p, long long n) {
    long long i = (long long)blockIdx.x * blockDim.x + threadIdx.x;
    if (i < n) p[i] = 0.0f;
}

// ---------------------------------------------------------------------------
// Degree accumulation: deg[dst[e]] += 1  (one thread per edge)
// ---------------------------------------------------------------------------
__global__ void k_degree(const int* __restrict__ dst, float* __restrict__ deg, int nE) {
    int e = blockIdx.x * blockDim.x + threadIdx.x;
    if (e < nE) atomicAdd(&deg[dst[e]], 1.0f);
}

// dinv[i] = rsqrt(deg[i] + 1)   (in place on the deg buffer)
__global__ void k_dinv(float* __restrict__ deg, int n) {
    int i = blockIdx.x * blockDim.x + threadIdx.x;
    if (i < n) deg[i] = __frsqrt_rn(deg[i] + 1.0f);
}

// ---------------------------------------------------------------------------
// WMMA GEMM: C[M,32] = A[M,K] @ W[K,32], f32 in, f16 WMMA, f32 accumulate.
// One wave -> one 16-row M tile, both 16-col N tiles. K in {32,128}.
//
// A fragment (16-bit A 16x32, ISA 7.12.2): row = lane&15;
//   lane<16  -> K = kk+0..15 (contiguous like B, not the ISA table order)
//   lane>=16 -> K = kk+16..31
// B fragment (f16 KxN): lanes 0-15 hold N=lane for K=kk+0..15 across elements,
//   lanes 16-31 hold N=lane-16 for K=kk+16..31.
// C/D: element r -> M = r + 8*(lane>>4), N = lane&15.
//
// W is staged into LDS *transposed* (sWt[n*K + k]) so each lane's 16-element
// B fragment is 32 contiguous, 32B-aligned bytes -> 2x ds_load_b128.
// ---------------------------------------------------------------------------
template <int K>
__global__ void k_gemm_wmma(const float* __restrict__ A,
                            const float* __restrict__ W,  // [K,32] row-major
                            float* __restrict__ C, int M) {
    __shared__ _Float16 sWt[32 * K];  // [n][k]
    for (int i = threadIdx.x; i < 32 * K; i += blockDim.x) {
        int n = i / K;   // K is a power of two -> shift
        int k = i % K;
        sWt[i] = (_Float16)W[k * 32 + n];
    }
    __syncthreads();

    const int lane = threadIdx.x & 31;
    const int wave = threadIdx.x >> 5;
    const int half = lane >> 4;
    const int l15  = lane & 15;
    const int m0   = (blockIdx.x * (blockDim.x >> 5) + wave) * 16;
    if (m0 >= M) return;             // m0 is wave-uniform -> EXEC all-1s for WMMA

    const long long row = m0 + l15;
    v8f acc0 = {};
    v8f acc1 = {};

    for (int kk = 0; kk < K; kk += 32) {
        // --- A fragment: 16 f16 per lane, one contiguous 16-float run (K=kk+16*half+0..15) ---
        const float* ap = A + row * K + kk + 16 * half;
        v4f a0 = *(const v4f*)(ap + 0);
        v4f a1 = *(const v4f*)(ap + 4);
        v4f a2 = *(const v4f*)(ap + 8);
        v4f a3 = *(const v4f*)(ap + 12);
        v16h a;
#pragma unroll
        for (int i = 0; i < 4; ++i) {
            a[i]      = (_Float16)a0[i];
            a[4 + i]  = (_Float16)a1[i];
            a[8 + i]  = (_Float16)a2[i];
            a[12 + i] = (_Float16)a3[i];
        }
        // --- B fragments: 16 contiguous f16 (32B aligned) per lane per tile ---
        const int kb = kk + 16 * half;
        v16h b0 = *(const v16h*)&sWt[(l15)      * K + kb];
        v16h b1 = *(const v16h*)&sWt[(16 + l15) * K + kb];

        acc0 = __builtin_amdgcn_wmma_f32_16x16x32_f16(false, a, false, b0,
                                                      (short)0, acc0, false, false);
        acc1 = __builtin_amdgcn_wmma_f32_16x16x32_f16(false, a, false, b1,
                                                      (short)0, acc1, false, false);
    }

#pragma unroll
    for (int r = 0; r < 8; ++r) {
        long long m = m0 + r + 8 * half;
        C[m * 32 + l15]      = acc0[r];
        C[m * 32 + 16 + l15] = acc1[r];
    }
}

// ---------------------------------------------------------------------------
// Edge scatter: agg[dst] += hx[src] * dinv[src]*dinv[dst]
// One wave per edge, one lane per channel: src/dst/dinv loads are wave-uniform
// broadcasts; hx gather and agg atomics are coalesced 128B segments in L2.
// ---------------------------------------------------------------------------
__global__ void k_scatter(const int* __restrict__ src, const int* __restrict__ dst,
                          const float* __restrict__ dinv,
                          const float* __restrict__ hx, float* __restrict__ agg,
                          int nE) {
    long long t = (long long)blockIdx.x * blockDim.x + threadIdx.x;
    int       e = (int)(t >> 5);
    int       c = (int)(t & 31);
    if (e >= nE) return;
    int   s = src[e];
    int   d = dst[e];
    float w = dinv[s] * dinv[d];
    atomicAdd(&agg[(long long)d * 32 + c], hx[(long long)s * 32 + c] * w);
}

// h[i,c] = relu(agg[i,c] + hx[i,c]*dinv[i]^2 + b[c])
__global__ void k_self_bias_relu(const float* __restrict__ agg,
                                 const float* __restrict__ hx,
                                 const float* __restrict__ dinv,
                                 const float* __restrict__ b,
                                 float* __restrict__ h, int n) {
    long long t = (long long)blockIdx.x * blockDim.x + threadIdx.x;
    int i = (int)(t >> 5);
    int c = (int)(t & 31);
    if (i >= n) return;
    float di = dinv[i];
    float v  = agg[(long long)i * 32 + c] + hx[(long long)i * 32 + c] * di * di + b[c];
    h[(long long)i * 32 + c] = fmaxf(v, 0.0f);
}

// ---------------------------------------------------------------------------
// Global mean pool: pooled[g] += h[i], cnt[g] += 1
// ---------------------------------------------------------------------------
__global__ void k_pool(const int* __restrict__ batch, const float* __restrict__ h,
                       float* __restrict__ pooled, float* __restrict__ cnt, int n) {
    long long t = (long long)blockIdx.x * blockDim.x + threadIdx.x;
    int i = (int)(t >> 5);
    int c = (int)(t & 31);
    if (i >= n) return;
    int g = batch[i];
    atomicAdd(&pooled[(long long)g * 32 + c], h[(long long)i * 32 + c]);
    if (c == 0) atomicAdd(&cnt[g], 1.0f);
}

// out[g,o] = (pooled[g,:] / max(cnt,1)) @ Wl[:,o] + bl[o]     (Wl is [32,2])
__global__ void k_head(const float* __restrict__ pooled, const float* __restrict__ cnt,
                       const float* __restrict__ Wl, const float* __restrict__ bl,
                       float* __restrict__ out, int nG) {
    int g = blockIdx.x * blockDim.x + threadIdx.x;
    if (g >= nG) return;
    float inv = 1.0f / fmaxf(cnt[g], 1.0f);
    float o0 = 0.0f, o1 = 0.0f;
#pragma unroll
    for (int c = 0; c < 32; ++c) {
        float p = pooled[g * 32 + c] * inv;
        o0 += p * Wl[c * 2 + 0];
        o1 += p * Wl[c * 2 + 1];
    }
    out[g * 2 + 0] = o0 + bl[0];
    out[g * 2 + 1] = o1 + bl[1];
}

// ---------------------------------------------------------------------------
extern "C" void kernel_launch(void* const* d_in, const int* in_sizes, int n_in,
                              void* d_out, int out_size, void* d_ws, size_t ws_size,
                              hipStream_t stream) {
    const float* x   = (const float*)d_in[0];
    const int*   ei  = (const int*)d_in[1];   // [2, E]: src then dst
    const int*   bat = (const int*)d_in[2];
    const float* W1  = (const float*)d_in[3];
    const float* b1  = (const float*)d_in[4];
    const float* W2  = (const float*)d_in[5];
    const float* b2  = (const float*)d_in[6];
    const float* Wl  = (const float*)d_in[7];
    const float* bl  = (const float*)d_in[8];
    float*       out = (float*)d_out;

    const int* src = ei;
    const int* dst = ei + N_EDGES;

    // Workspace layout (floats)
    float* ws     = (float*)d_ws;
    float* dinv   = ws;                                   // N (deg, then rsqrt)
    float* hx     = dinv + N_NODES;                       // N*32
    float* agg    = hx + (long long)N_NODES * 32;         // N*32
    float* h      = agg + (long long)N_NODES * 32;        // N*32
    float* pooled = h + (long long)N_NODES * 32;          // G*32
    float* cnt    = pooled + (long long)N_GRAPHS * 32;    // G

    const int B = 256;
    const long long featN = (long long)N_NODES * 32;
    const long long edgeT = (long long)N_EDGES * 32;
    dim3 gFeat((unsigned)((featN + B - 1) / B));
    dim3 gEdgeT((unsigned)((edgeT + B - 1) / B));
    dim3 gEdge((N_EDGES + B - 1) / B);
    dim3 gNode((N_NODES + B - 1) / B);
    // GEMM: 256 threads = 8 waves, 8 M-tiles of 16 rows per block
    dim3 gGemm((N_NODES / 16 + 8 - 1) / 8);

    // --- normalization ---
    k_zero<<<gNode, B, 0, stream>>>(dinv, N_NODES);
    k_degree<<<gEdge, B, 0, stream>>>(dst, dinv, N_EDGES);
    k_dinv<<<gNode, B, 0, stream>>>(dinv, N_NODES);

    // --- layer 1 ---
    k_gemm_wmma<128><<<gGemm, B, 0, stream>>>(x, W1, hx, N_NODES);
    k_zero<<<gFeat, B, 0, stream>>>(agg, featN);
    k_scatter<<<gEdgeT, B, 0, stream>>>(src, dst, dinv, hx, agg, N_EDGES);
    k_self_bias_relu<<<gFeat, B, 0, stream>>>(agg, hx, dinv, b1, h, N_NODES);

    // --- layer 2 ---
    k_gemm_wmma<32><<<gGemm, B, 0, stream>>>(h, W2, hx, N_NODES);
    k_zero<<<gFeat, B, 0, stream>>>(agg, featN);
    k_scatter<<<gEdgeT, B, 0, stream>>>(src, dst, dinv, hx, agg, N_EDGES);
    k_self_bias_relu<<<gFeat, B, 0, stream>>>(agg, hx, dinv, b2, h, N_NODES);

    // --- pool + head ---
    k_zero<<<dim3((N_GRAPHS * 32 + 512 + B - 1) / B), B, 0, stream>>>(pooled,
                                                       N_GRAPHS * 32 + N_GRAPHS);
    k_pool<<<gFeat, B, 0, stream>>>(bat, h, pooled, cnt, N_NODES);
    k_head<<<dim3((N_GRAPHS + B - 1) / B), B, 0, stream>>>(pooled, cnt, Wl, bl,
                                                           out, N_GRAPHS);
}
